// Model_56556129354475
// MI455X (gfx1250) — matrix operand order repaired and checked by reading, not compiled, not run on hardware
//
#include <hip/hip_runtime.h>

// ---------------------------------------------------------------------------
// APPNP-style GNN:  h1 = feats@W1+b1 ; h = relu(h1)@W2+b2 ; K rounds of
// normalized gather/scatter propagation.
// GEMMs: v_wmma_f32_16x16x32_bf16 with bf16 hi/lo split (3 WMMAs / K=32) for
// ~fp32 accuracy.  B fragments are staged into LDS with double-buffered
// global_load_async_to_lds_b128 (ASYNCcnt) and consumed via ds_load_b128,
// cutting per-wave L2 refetch of B by 8x per block.
// ---------------------------------------------------------------------------

#define N_NODES 100000
#define N_EDGES 1600000
#define IN_DIM  512
#define HID_DIM 256
#define OUT_DIM 64
#define K_ITERS 10
#define ALPHA   0.1f

typedef __attribute__((ext_vector_type(16))) __bf16 v16bf;
typedef __attribute__((ext_vector_type(8)))  float  v8f;

// 16B async copy, GVS form: VDST = LDS byte offset, VADDR = 32-bit byte
// offset, SADDR = 64-bit base (SGPR pair).  Tracked by ASYNCcnt.
__device__ __forceinline__ void async_g2l_b128(unsigned int lds_off,
                                               unsigned int gbyte_off,
                                               unsigned long long sbase) {
    asm volatile("global_load_async_to_lds_b128 %0, %1, %2"
                 :: "v"(lds_off), "v"(gbyte_off), "s"(sbase)
                 : "memory");
}

__device__ __forceinline__ void wait_async0() {
    asm volatile("s_wait_asynccnt 0x0" ::: "memory");
}

// ---------------------------------------------------------------------------
// Pack a row-major [kdim, ncol] fp32 weight matrix into WMMA B-fragment order,
// pre-split into bf16 hi/lo.  Fragment layout (16x16x32 bf16 B = 32 x 16):
//   lanes 0-15 : N = lane,     halves e=0..7 hold K = ks*32 + e, e=8..15 hold K = ks*32 + 16 + (e-8)
//   lanes 16-31: N = lane-16,  halves e=0..7 hold K = ks*32 + 8 + e, e=8..15 hold K = ks*32 + 24 + (e-8)
// Linear index: (((ntile*(kdim/32) + ks)*32 + lane)*16 + e)
// ---------------------------------------------------------------------------
__global__ void pack_w_kernel(const float* __restrict__ W,
                              __bf16* __restrict__ Whi,
                              __bf16* __restrict__ Wlo,
                              int kdim, int ncol) {
    int idx = blockIdx.x * blockDim.x + threadIdx.x;
    int total = kdim * ncol;
    if (idx >= total) return;
    int e      = idx & 15;
    int lane   = (idx >> 4) & 31;
    int rem    = idx >> 9;              // ntile*(kdim/32) + ks
    int ksteps = kdim >> 5;
    int ks     = rem % ksteps;
    int ntile  = rem / ksteps;
    int k   = ks * 32 + ((lane >> 4) << 3) + (e & 7) + ((e >> 3) << 4);
    int col = ntile * 16 + (lane & 15);
    float v = W[(size_t)k * ncol + col];
    __bf16 hi = (__bf16)v;
    __bf16 lo = (__bf16)(v - (float)hi);
    Whi[idx] = hi;
    Wlo[idx] = lo;
}

// ---------------------------------------------------------------------------
// WMMA GEMM:  out[M,NCOL] = act(A[M,KDIM]) @ B[KDIM,NCOL] + bias
// Block = 256 threads = 8 waves.  One wave: 16-row M-tile x NT=4 N-tiles.
// WPM = NCOL/64 waves per M-tile; a block covers 8/WPM M-tiles and uses ALL
// NCOL/16 n-tiles of B -> one shared LDS B-slice per K-step per block.
// Per K-step B slice = (NCOL/16) KB per hi/lo half; double-buffered in LDS.
// ---------------------------------------------------------------------------
template <int KDIM, int NCOL, bool RELU_IN>
__global__ __launch_bounds__(256)
void gemm_wmma_kernel(const float* __restrict__ A,
                      const __bf16* __restrict__ Bhi,
                      const __bf16* __restrict__ Blo,
                      const float* __restrict__ bias,
                      float* __restrict__ out,
                      float* __restrict__ out2) {
    constexpr int NT      = 4;
    constexpr int KSTEPS  = KDIM / 32;
    constexpr int WPM     = NCOL / (16 * NT);       // waves per M-tile
    constexpr int MTILES  = N_NODES / 16;
    constexpr int NTILESB = NCOL / 16;              // n-tiles staged per block
    constexpr int CHUNK   = NTILESB * 1024;         // bytes per half (hi|lo)
    constexpr int BUFB    = CHUNK * 2;              // bytes per buffer
    constexpr int UNITS   = BUFB / 16;              // 16B copy units per buffer
    constexpr int UPT     = UNITS / 256;            // units per thread

    __shared__ __align__(16) unsigned char smem[2][BUFB];

    const int lane  = threadIdx.x & 31;
    const int wave  = threadIdx.x >> 5;
    const int mtile = blockIdx.x * (8 / WPM) + wave / WPM;
    const int ngrp  = wave % WPM;
    const bool active = (mtile < MTILES);           // wave-uniform

    const unsigned long long hiBase = (unsigned long long)Bhi;
    const unsigned long long loBase = (unsigned long long)Blo;
    const unsigned int smemBase = (unsigned int)(size_t)(&smem[0][0]);

    // async copy of one K-step B slice (hi+lo) into buffer b; all 256 threads
    auto issueB = [&](int ks, int b) {
#pragma unroll
        for (int i = 0; i < UPT; ++i) {
            const int u    = threadIdx.x + i * 256;      // 0..UNITS-1
            const int half = (u >= UNITS / 2) ? 1 : 0;   // 0=hi, 1=lo
            const int uu   = u - half * (UNITS / 2);
            const int nt   = uu >> 6;                    // n-tile
            const int rem  = uu & 63;                    // 16B unit in tile
            const unsigned int goff =
                (unsigned int)(((nt * KSTEPS + ks) << 10) + (rem << 4)); // bytes
            const unsigned int loff =
                smemBase + (unsigned int)(b * BUFB + half * CHUNK + nt * 1024 + rem * 16);
            async_g2l_b128(loff, goff, half ? loBase : hiBase);
        }
    };

    const int m  = mtile * 16 + (lane & 15);
    const int kb = (lane >> 4) * 8;
    const float* __restrict__ arow = A + (size_t)(active ? m : 0) * KDIM;

    v8f c[NT];
    const v8f zero = {};
#pragma unroll
    for (int t = 0; t < NT; ++t) c[t] = zero;

    issueB(0, 0);

    for (int ks = 0; ks < KSTEPS; ++ks) {
        const int cur = ks & 1;
        wait_async0();           // my copies for smem[cur] have landed
        __syncthreads();         // everyone's copies landed; prev buffer consumed
        if (ks + 1 < KSTEPS) issueB(ks + 1, 1 - cur);   // overlap with compute

        if (active) {
            // A fragment (16x32 bf16): lane L holds row M=(L&15); halves 0..7
            // hold K=kb..kb+7, halves 8..15 hold K=kb+16..kb+23, kb=(L>=16)*8.
            const int kbase = ks * 32 + kb;
            const float4 f0 = *(const float4*)(arow + kbase);
            const float4 f1 = *(const float4*)(arow + kbase + 4);
            const float4 f2 = *(const float4*)(arow + kbase + 16);
            const float4 f3 = *(const float4*)(arow + kbase + 20);
            float fa[16] = {f0.x, f0.y, f0.z, f0.w, f1.x, f1.y, f1.z, f1.w,
                            f2.x, f2.y, f2.z, f2.w, f3.x, f3.y, f3.z, f3.w};
            if (RELU_IN) {
#pragma unroll
                for (int e = 0; e < 16; ++e) fa[e] = fmaxf(fa[e], 0.0f);
            }
            v16bf ah, al;
#pragma unroll
            for (int e = 0; e < 16; ++e) {
                __bf16 h = (__bf16)fa[e];
                ah[e] = h;
                al[e] = (__bf16)(fa[e] - (float)h);
            }
#pragma unroll
            for (int t = 0; t < NT; ++t) {
                const int nt = ngrp * NT + t;
                const v16bf bh = *(const v16bf*)&smem[cur][nt * 1024 + lane * 32];
                const v16bf bl = *(const v16bf*)&smem[cur][CHUNK + nt * 1024 + lane * 32];
                // bf16x3 fp32 emulation: AhiBhi + AloBhi + AhiBlo
                c[t] = __builtin_amdgcn_wmma_f32_16x16x32_bf16(
                           false, ah, false, bh, (short)0, c[t], false, false);
                c[t] = __builtin_amdgcn_wmma_f32_16x16x32_bf16(
                           false, al, false, bh, (short)0, c[t], false, false);
                c[t] = __builtin_amdgcn_wmma_f32_16x16x32_bf16(
                           false, ah, false, bl, (short)0, c[t], false, false);
            }
        }
        __syncthreads();         // all waves done reading smem[cur]
    }

    if (!active) return;

    // C/D layout: lane L -> col (L&15); VGPR r -> row r + (L>=16)*8
    const int mrow0 = mtile * 16 + ((lane >> 4) << 3);
#pragma unroll
    for (int t = 0; t < NT; ++t) {
        const int col = (ngrp * NT + t) * 16 + (lane & 15);
        const float bv = bias[col];
#pragma unroll
        for (int r = 0; r < 8; ++r) {
            const float v = c[t][r] + bv;
            const size_t o = (size_t)(mrow0 + r) * NCOL + col;
            out[o] = v;
            if (out2) out2[o] = v;
        }
    }
}

// ---------------------------------------------------------------------------
// Propagation support kernels
// ---------------------------------------------------------------------------
__global__ void degree_kernel(const int* __restrict__ src, const int* __restrict__ dst,
                              float* __restrict__ deg_out, float* __restrict__ deg_in,
                              int nE) {
    int e = blockIdx.x * blockDim.x + threadIdx.x;
    if (e >= nE) return;
    atomicAdd(deg_out + src[e], 1.0f);
    atomicAdd(deg_in  + dst[e], 1.0f);
}

__global__ void edge_weight_kernel(const int* __restrict__ src, const int* __restrict__ dst,
                                   const float* __restrict__ deg_out,
                                   const float* __restrict__ deg_in,
                                   float* __restrict__ w, int nE) {
    int e = blockIdx.x * blockDim.x + threadIdx.x;
    if (e >= nE) return;
    w[e] = rsqrtf(fmaxf(deg_out[src[e]], 1.0f)) * rsqrtf(fmaxf(deg_in[dst[e]], 1.0f));
}

// 64 threads per edge; a wave32 covers one contiguous half of an edge's 64 dims
// -> fully coalesced gather loads and coalesced fp32 atomic scatter.
__global__ void scatter_kernel(const float* __restrict__ h,
                               const int* __restrict__ src, const int* __restrict__ dst,
                               const float* __restrict__ w,
                               float* __restrict__ agg) {
    const long long gid = (long long)blockIdx.x * blockDim.x + threadIdx.x;
    const int e = (int)(gid >> 6);
    const int d = (int)(gid & 63);
    if (e >= N_EDGES) return;
    const float v = h[(size_t)src[e] * OUT_DIM + d] * w[e];
    atomicAdd(agg + (size_t)dst[e] * OUT_DIM + d, v);
}

__global__ void combine_kernel(const float* __restrict__ agg, const float* __restrict__ h0,
                               float* __restrict__ hout, int n) {
    int i = blockIdx.x * blockDim.x + threadIdx.x;
    if (i >= n) return;
    hout[i] = (1.0f - ALPHA) * agg[i] + ALPHA * h0[i];
}

// ---------------------------------------------------------------------------
// Launch
// ---------------------------------------------------------------------------
extern "C" void kernel_launch(void* const* d_in, const int* in_sizes, int n_in,
                              void* d_out, int out_size, void* d_ws, size_t ws_size,
                              hipStream_t stream) {
    const float* feats = (const float*)d_in[0];
    const int*   src   = (const int*)d_in[1];
    const int*   dst   = (const int*)d_in[2];
    const float* W1    = (const float*)d_in[3];
    const float* b1    = (const float*)d_in[4];
    const float* W2    = (const float*)d_in[5];
    const float* b2    = (const float*)d_in[6];

    float* h1_out = (float*)d_out;                               // [N, 256]
    float* h_cur  = h1_out + (size_t)N_NODES * HID_DIM;          // [N, 64] final output

    char* ws = (char*)d_ws;
    float* deg_out = (float*)ws;  ws += (size_t)N_NODES * 4;
    float* deg_in  = (float*)ws;  ws += (size_t)N_NODES * 4;
    float* wedge   = (float*)ws;  ws += (size_t)N_EDGES * 4;
    float* h0      = (float*)ws;  ws += (size_t)N_NODES * OUT_DIM * 4;
    float* agg     = (float*)ws;  ws += (size_t)N_NODES * OUT_DIM * 4;
    __bf16* W1hi = (__bf16*)ws;   ws += (size_t)IN_DIM * HID_DIM * 2;
    __bf16* W1lo = (__bf16*)ws;   ws += (size_t)IN_DIM * HID_DIM * 2;
    __bf16* W2hi = (__bf16*)ws;   ws += (size_t)HID_DIM * OUT_DIM * 2;
    __bf16* W2lo = (__bf16*)ws;   ws += (size_t)HID_DIM * OUT_DIM * 2;

    // degrees must start at zero (deg_out/deg_in are contiguous)
    hipMemsetAsync(deg_out, 0, (size_t)2 * N_NODES * 4, stream);

    // weight packing (tiny)
    pack_w_kernel<<<(IN_DIM * HID_DIM + 255) / 256, 256, 0, stream>>>(
        W1, W1hi, W1lo, IN_DIM, HID_DIM);
    pack_w_kernel<<<(HID_DIM * OUT_DIM + 255) / 256, 256, 0, stream>>>(
        W2, W2hi, W2lo, HID_DIM, OUT_DIM);

    // symmetric-normalization edge weights
    degree_kernel<<<(N_EDGES + 255) / 256, 256, 0, stream>>>(src, dst, deg_out, deg_in, N_EDGES);
    edge_weight_kernel<<<(N_EDGES + 255) / 256, 256, 0, stream>>>(
        src, dst, deg_out, deg_in, wedge, N_EDGES);

    // GEMM1: h1 = feats @ W1 + b1   (WPM=4 -> 2 M-tiles/block, 64KB LDS staged B)
    gemm_wmma_kernel<IN_DIM, HID_DIM, false><<<(N_NODES / 16) * 4 / 8, 256, 0, stream>>>(
        feats, W1hi, W1lo, b1, h1_out, nullptr);

    // GEMM2: h0 = relu(h1) @ W2 + b2 (WPM=1 -> 8 M-tiles/block); also seeds h_cur
    gemm_wmma_kernel<HID_DIM, OUT_DIM, true><<<(N_NODES / 16 + 7) / 8, 256, 0, stream>>>(
        h1_out, W2hi, W2lo, b2, h_cur, h0);

    // K rounds of personalized-PageRank propagation
    const int scatter_blocks = (int)(((long long)N_EDGES * OUT_DIM) / 256);
    const int combine_n      = N_NODES * OUT_DIM;
    for (int it = 0; it < K_ITERS; ++it) {
        hipMemsetAsync(agg, 0, (size_t)N_NODES * OUT_DIM * 4, stream);
        scatter_kernel<<<scatter_blocks, 256, 0, stream>>>(h_cur, src, dst, wedge, agg);
        combine_kernel<<<(combine_n + 255) / 256, 256, 0, stream>>>(agg, h0, h_cur, combine_n);
    }
}
